// ToxicityGATv2_75462575390874
// MI455X (gfx1250) — matrix operand order repaired and checked by reading, not compiled, not run on hardware
//
#include <hip/hip_runtime.h>
#include <hip/hip_bf16.h>
#include <math.h>

// ---------------------------------------------------------------------------
// ToxicityGATv2 pipeline for gfx1250 (MI455X).
// Dense GEMMs use v_wmma_f32_16x16x32_bf16 (bf16 A/B, f32 accumulate).
// Edge phase uses coalesced loads + global_atomic_add_f32 and order-preserving
// uint atomicMax for segment-max. All pre-BN biases cancel analytically.
// ---------------------------------------------------------------------------

typedef __attribute__((ext_vector_type(16))) __bf16 v16bf;
typedef __attribute__((ext_vector_type(8)))  __bf16 v8bf;
typedef __attribute__((ext_vector_type(8)))  float  v8f;

#define TPB 256
#define CDIV(a, b) (((a) + (b) - 1) / (b))
#define ENC_NEG_INF 0x007FFFFFu  // encf(-inf)

__device__ __forceinline__ unsigned encf(float f) {
  unsigned u = __float_as_uint(f);
  return (u & 0x80000000u) ? ~u : (u | 0x80000000u);
}
__device__ __forceinline__ float decf(unsigned x) {
  return (x & 0x80000000u) ? __uint_as_float(x ^ 0x80000000u)
                           : __uint_as_float(~x);
}

// -------------------------- generic small kernels --------------------------

__global__ void k_init_u32(unsigned* __restrict__ p, unsigned v, int n) {
  int i = blockIdx.x * TPB + threadIdx.x;
  if (i < n) p[i] = v;
}

// Column mean / inv-std over rows (training-mode BatchNorm statistics).
__global__ void k_col_stats(const float* __restrict__ X, int rows, int cols,
                            float* __restrict__ mean, float* __restrict__ istd) {
  int c = blockIdx.x;
  float s = 0.f, q = 0.f;
  for (int r = threadIdx.x; r < rows; r += TPB) {
    float v = X[(size_t)r * cols + c];
    s += v; q += v * v;
  }
  __shared__ float ss[TPB], qq[TPB];
  ss[threadIdx.x] = s; qq[threadIdx.x] = q;
  __syncthreads();
  for (int o = TPB / 2; o > 0; o >>= 1) {
    if (threadIdx.x < o) { ss[threadIdx.x] += ss[threadIdx.x + o]; qq[threadIdx.x] += qq[threadIdx.x + o]; }
    __syncthreads();
  }
  if (threadIdx.x == 0) {
    float m = ss[0] / (float)rows;
    float v = qq[0] / (float)rows - m * m;
    mean[c] = m;
    istd[c] = rsqrtf(fmaxf(v, 0.f) + 1e-5f);
  }
}

// ------------------------------ WMMA GEMM ----------------------------------
// Pack f32 weight matrix B[K, NC] (row-major) into the per-lane bf16 layout
// of V_WMMA_F32_16X16X32_BF16's B operand: for each (ktile, ctile, lane),
// 16 bf16 = K values kbase(lane)+0..7 then kbase(lane)+16..23 of column col(lane).
__global__ void k_pack_b(const float* __restrict__ B, __bf16* __restrict__ Bp,
                         int K, int NC) {
  int idx = blockIdx.x * TPB + threadIdx.x;
  int total = K * NC;  // (K/32)*(NC/16)*32*16
  if (idx >= total) return;
  int i = idx & 15;
  int L = (idx >> 4) & 31;
  int t = idx >> 9;  // kt*ctiles + ct
  int ctiles = NC >> 4;
  int ct = t % ctiles;
  int kt = t / ctiles;
  int col = ct * 16 + (L & 15);
  int kbase = (L < 16) ? 0 : 8;
  int k = kt * 32 + kbase + i + (i & 8);
  Bp[idx] = (__bf16)B[(size_t)k * NC + col];
}

// C[M, NC] = A[M, K] (bf16, row-major) @ Bpacked. One wave -> one 16x16 tile.
__global__ void k_gemm_wmma(const __bf16* __restrict__ A,
                            const __bf16* __restrict__ Bp,
                            float* __restrict__ C, int M, int K, int NC) {
  int ct = blockIdx.x;
  int mt = blockIdx.y;
  int L = threadIdx.x;  // 0..31 (wave32)
  int ctiles = NC >> 4;
  int row = mt * 16 + (L & 15);
  if (row >= M) row = M - 1;  // clamped load for ragged M (stores guarded)
  int kb = (L < 16) ? 0 : 8;  // A-layout: lanes 0-15 K 0..7/16..23; 16-31 K 8..15/24..31
  const __bf16* Arow = A + (size_t)row * K;
  v8f acc = {};
  int ktiles = K >> 5;
  for (int kt = 0; kt < ktiles; ++kt) {
    int k0 = kt * 32;
    v8bf a0 = *(const v8bf*)(Arow + k0 + kb);
    v8bf a1 = *(const v8bf*)(Arow + k0 + kb + 16);
    v16bf a;
#pragma unroll
    for (int i = 0; i < 8; ++i) { a[i] = a0[i]; a[i + 8] = a1[i]; }
    v16bf b = *(const v16bf*)(Bp + (((size_t)kt * ctiles + ct) * 32 + L) * 16);
    acc = __builtin_amdgcn_wmma_f32_16x16x32_bf16(
        /*neg_a=*/false, a, /*neg_b=*/false, b,
        /*c_mod=*/(short)0, acc, /*reuse_a=*/false, /*reuse_b=*/false);
  }
  int col = ct * 16 + (L & 15);
  int rbase = mt * 16 + ((L < 16) ? 0 : 8);
#pragma unroll
  for (int j = 0; j < 8; ++j) {
    int r = rbase + j;
    if (r < M) C[(size_t)r * NC + col] = acc[j];
  }
}

// ----------------------------- input layer ---------------------------------
// H[n, :] = x[n, :39] @ in_W[39, 128]   (in_b cancels in BN)
__global__ void k_in_gemm(const float* __restrict__ x,
                          const float* __restrict__ W,
                          float* __restrict__ H) {
  __shared__ float xs[40];
  int n = blockIdx.x;
  int c = threadIdx.x;  // 0..127
  if (c < 39) xs[c] = x[(size_t)n * 39 + c];
  __syncthreads();
  float s = 0.f;
#pragma unroll
  for (int k = 0; k < 39; ++k) s += xs[k] * W[k * 128 + c];
  H[(size_t)n * 128 + c] = s;
}

__global__ void k_bn_relu(const float* __restrict__ H, const float* __restrict__ mean,
                          const float* __restrict__ istd, const float* __restrict__ g,
                          const float* __restrict__ beta, float* __restrict__ xf,
                          __bf16* __restrict__ xb, int total, int cols) {
  int i = blockIdx.x * TPB + threadIdx.x;
  if (i >= total) return;
  int c = i % cols;
  float v = g[c] * (H[i] - mean[c]) * istd[c] + beta[c];
  v = fmaxf(v, 0.f);
  if (xf) xf[i] = v;
  xb[i] = (__bf16)v;
}

// BN -> ELU -> residual add into xn, refresh bf16 mirror.
__global__ void k_bn_elu_res(const float* __restrict__ AGG, const float* __restrict__ mean,
                             const float* __restrict__ istd, const float* __restrict__ g,
                             const float* __restrict__ beta, float* __restrict__ xn,
                             __bf16* __restrict__ xnbf, int total) {
  int i = blockIdx.x * TPB + threadIdx.x;
  if (i >= total) return;
  int c = i & 127;
  float v = g[c] * (AGG[i] - mean[c]) * istd[c] + beta[c];
  v = (v > 0.f) ? v : expm1f(v);
  float o = xn[i] + v;
  xn[i] = o;
  xnbf[i] = (__bf16)o;
}

// ------------------------------ edge phase ---------------------------------
// alpha[e,h] = sum_d lrelu(xl[src]+xr[dst]+ea@We)[h,d] * att[h,d]; seg-max via
// order-preserving uint atomicMax. We (8x128) + att (128) cached in LDS.
__global__ void k_edge_alpha(const float* __restrict__ xl, const float* __restrict__ xr,
                             const float* __restrict__ ea, const int* __restrict__ src,
                             const int* __restrict__ dst, const float* __restrict__ We,
                             const float* __restrict__ att, float* __restrict__ alpha,
                             unsigned* __restrict__ amaxb, int E_) {
  __shared__ float WeS[8 * 128];
  __shared__ float attS[128];
  for (int i = threadIdx.x; i < 8 * 128; i += TPB) WeS[i] = We[i];
  if (threadIdx.x < 128) attS[threadIdx.x] = att[threadIdx.x];
  __syncthreads();
  int gid = blockIdx.x * TPB + threadIdx.x;
  if (gid >= E_ * 4) return;
  int e = gid >> 2, h = gid & 3;
  int s = src[e], d = dst[e];
  float eav[8];
#pragma unroll
  for (int k = 0; k < 8; ++k) eav[k] = ea[(size_t)e * 8 + k];
  const float* xls = xl + (size_t)s * 128 + h * 32;
  const float* xrd = xr + (size_t)d * 128 + h * 32;
  float al = 0.f;
#pragma unroll 4
  for (int dd = 0; dd < 32; ++dd) {
    int cc = h * 32 + dd;
    float ep = 0.f;
#pragma unroll
    for (int k = 0; k < 8; ++k) ep += eav[k] * WeS[k * 128 + cc];
    float m = xls[dd] + xrd[dd] + ep;
    m = (m > 0.f) ? m : 0.2f * m;
    al += m * attS[cc];
  }
  alpha[gid] = al;
  atomicMax(&amaxb[d * 4 + h], encf(al));
}

__global__ void k_amax_fix(const unsigned* __restrict__ amaxb,
                           float* __restrict__ amax, int n) {
  int i = blockIdx.x * TPB + threadIdx.x;
  if (i >= n) return;
  float v = decf(amaxb[i]);
  amax[i] = __builtin_isfinite(v) ? v : 0.f;  // nodes with no in-edges
}

// alpha <- exp(alpha - amax[dst]); denom[dst] += alpha
__global__ void k_edge_den(float* __restrict__ alpha, const int* __restrict__ dst,
                           const float* __restrict__ amax, float* __restrict__ denom,
                           int E_) {
  int gid = blockIdx.x * TPB + threadIdx.x;
  if (gid >= E_ * 4) return;
  int e = gid >> 2, h = gid & 3;
  int d = dst[e];
  float ae = expf(alpha[gid] - amax[d * 4 + h]);
  alpha[gid] = ae;
  atomicAdd(&denom[d * 4 + h], ae);
}

// agg[dst, h*32+d] += xl[src, h*32+d] * (ae / (denom+eps))
__global__ void k_edge_scatter(const float* __restrict__ alpha, const int* __restrict__ src,
                               const int* __restrict__ dst, const float* __restrict__ denom,
                               const float* __restrict__ xl, float* __restrict__ agg, int E_) {
  int gid = blockIdx.x * TPB + threadIdx.x;
  if (gid >= E_ * 4) return;
  int e = gid >> 2, h = gid & 3;
  int s = src[e], d = dst[e];
  float a = alpha[gid] / (denom[d * 4 + h] + 1e-16f);
  const float* xls = xl + (size_t)s * 128 + h * 32;
  float* ag = agg + (size_t)d * 128 + h * 32;
#pragma unroll 8
  for (int dd = 0; dd < 32; ++dd) atomicAdd(&ag[dd], xls[dd] * a);
}

// ------------------------------- pooling -----------------------------------
__global__ void k_pool_scatter(const float* __restrict__ xn, const int* __restrict__ batch,
                               float* __restrict__ sadd, unsigned* __restrict__ smaxb,
                               float* __restrict__ cnt, int N_) {
  int i = blockIdx.x * TPB + threadIdx.x;
  if (i >= N_ * 128) return;
  int n = i >> 7, d = i & 127;
  int g = batch[n];
  float v = xn[i];
  atomicAdd(&sadd[(size_t)g * 128 + d], v);
  atomicMax(&smaxb[(size_t)g * 128 + d], encf(v));
  if (d == 0) atomicAdd(&cnt[g], 1.0f);
}

// xg = [mean | max | sum] per graph, emitted directly as bf16 GEMM operand.
__global__ void k_pool_fin(const float* __restrict__ sadd, const unsigned* __restrict__ smaxb,
                           const float* __restrict__ cnt, __bf16* __restrict__ xgbf, int G_) {
  int i = blockIdx.x * TPB + threadIdx.x;
  if (i >= G_ * 128) return;
  int g = i >> 7, d = i & 127;
  float c = cnt[g];
  float sa = sadd[i];
  float mx = decf(smaxb[i]);
  mx = (c > 0.f && __builtin_isfinite(mx)) ? mx : 0.f;
  float me = sa / fmaxf(c, 1.f);
  __bf16* row = xgbf + (size_t)g * 384;
  row[d] = (__bf16)me;
  row[128 + d] = (__bf16)mx;
  row[256 + d] = (__bf16)sa;
}

// ------------------------------- head --------------------------------------
__global__ void k_head(const float* __restrict__ t2, const float* __restrict__ W,
                       const float* __restrict__ b, float* __restrict__ out, int G_) {
  int i = blockIdx.x * TPB + threadIdx.x;
  if (i >= G_ * 12) return;
  int g = i / 12, c = i % 12;
  const float* tg = t2 + (size_t)g * 128;
  float s = b[c];
#pragma unroll 8
  for (int k = 0; k < 128; ++k) s += tg[k] * W[k * 12 + c];
  out[i] = 1.f / (1.f + expf(-s));
}

// ------------------------------- launcher ----------------------------------

extern "C" void kernel_launch(void* const* d_in, const int* in_sizes, int n_in,
                              void* d_out, int out_size, void* d_ws, size_t ws_size,
                              hipStream_t stream) {
  const float* x          = (const float*)d_in[0];
  const int*   edge_index = (const int*)d_in[1];
  const float* edge_attr  = (const float*)d_in[2];
  const int*   batch      = (const int*)d_in[3];
  const float* in_W       = (const float*)d_in[4];
  // d_in[5] in_b: cancels under training-mode BN
  const float* in_g       = (const float*)d_in[6];
  const float* in_beta    = (const float*)d_in[7];
  const float* Wl         = (const float*)d_in[8];
  const float* Wr         = (const float*)d_in[9];
  const float* We         = (const float*)d_in[10];
  const float* att        = (const float*)d_in[11];
  // d_in[12] conv_b: cancels under BN
  const float* bn_g       = (const float*)d_in[13];
  const float* bn_b       = (const float*)d_in[14];
  const float* t1_W       = (const float*)d_in[15];
  // d_in[16] t1_b: cancels under BN
  const float* t1_g       = (const float*)d_in[17];
  const float* t1_beta    = (const float*)d_in[18];
  const float* t2_W       = (const float*)d_in[19];
  // d_in[20] t2_b: cancels under BN
  const float* t2_g       = (const float*)d_in[21];
  const float* t2_beta    = (const float*)d_in[22];
  const float* head_W     = (const float*)d_in[23];
  const float* head_b     = (const float*)d_in[24];
  float* out = (float*)d_out;

  const int N_ = in_sizes[3];       // 50000
  const int E_ = in_sizes[2] / 8;   // 400000
  const int G_ = out_size / 12;     // 2000
  const int src_off = 0, dst_off = E_;
  const int* src = edge_index + src_off;
  const int* dst = edge_index + dst_off;

  // ---- workspace carve (256B aligned) ----
  char* base = (char*)d_ws;
  size_t off = 0;
  auto alloc = [&](size_t bytes) -> char* {
    char* r = base + off;
    off = (off + bytes + 255) & ~(size_t)255;
    return r;
  };
  float*    xn    = (float*)   alloc((size_t)N_ * 128 * 4);
  float*    xl    = (float*)   alloc((size_t)N_ * 128 * 4);
  float*    xr    = (float*)   alloc((size_t)N_ * 128 * 4);
  float*    agg   = (float*)   alloc((size_t)N_ * 128 * 4);   // also input pre-BN
  __bf16*   xnbf  = (__bf16*)  alloc((size_t)N_ * 128 * 2);
  float*    alpha = (float*)   alloc((size_t)E_ * 4 * 4);
  unsigned* amaxb = (unsigned*)alloc((size_t)N_ * 4 * 4);
  float*    amax  = (float*)   alloc((size_t)N_ * 4 * 4);
  float*    denom = (float*)   alloc((size_t)N_ * 4 * 4);
  float*    mean  = (float*)   alloc(256 * 4);
  float*    istd  = (float*)   alloc(256 * 4);
  __bf16*   packA = (__bf16*)  alloc((size_t)384 * 256 * 2);
  __bf16*   packB = (__bf16*)  alloc((size_t)384 * 256 * 2);
  float*    cnt   = (float*)   alloc((size_t)G_ * 4);
  float*    sadd  = (float*)   alloc((size_t)G_ * 128 * 4);
  unsigned* smaxb = (unsigned*)alloc((size_t)G_ * 128 * 4);
  __bf16*   xgbf  = (__bf16*)  alloc((size_t)G_ * 384 * 2);
  float*    t1pre = (float*)   alloc((size_t)G_ * 256 * 4);
  __bf16*   t1bf  = (__bf16*)  alloc((size_t)G_ * 256 * 2);
  float*    t2pre = (float*)   alloc((size_t)G_ * 128 * 4);
  float*    t2act = (float*)   alloc((size_t)G_ * 128 * 4);
  __bf16*   t2bf  = (__bf16*)  alloc((size_t)G_ * 128 * 2);
  (void)ws_size; (void)n_in;

  const int NH = N_ * 128;
  const int E4 = E_ * 4;

  // ---- input layer: h = x @ in_W ; BN ; ReLU ----
  k_in_gemm<<<N_, 128, 0, stream>>>(x, in_W, agg);
  k_col_stats<<<128, TPB, 0, stream>>>(agg, N_, 128, mean, istd);
  k_bn_relu<<<CDIV(NH, TPB), TPB, 0, stream>>>(agg, mean, istd, in_g, in_beta,
                                               xn, xnbf, NH, 128);

  // ---- 3 residual GATv2 layers ----
  for (int l = 0; l < 3; ++l) {
    const float* Wl_l  = Wl + (size_t)l * 128 * 128;
    const float* Wr_l  = Wr + (size_t)l * 128 * 128;
    const float* We_l  = We + (size_t)l * 8 * 128;
    const float* att_l = att + (size_t)l * 128;

    k_pack_b<<<CDIV(128 * 128, TPB), TPB, 0, stream>>>(Wl_l, packA, 128, 128);
    k_pack_b<<<CDIV(128 * 128, TPB), TPB, 0, stream>>>(Wr_l, packB, 128, 128);
    {
      dim3 grid(128 / 16, CDIV(N_, 16));
      k_gemm_wmma<<<grid, 32, 0, stream>>>(xnbf, packA, xl, N_, 128, 128);
      k_gemm_wmma<<<grid, 32, 0, stream>>>(xnbf, packB, xr, N_, 128, 128);
    }

    hipMemsetAsync(denom, 0, (size_t)N_ * 4 * 4, stream);
    hipMemsetAsync(agg, 0, (size_t)NH * 4, stream);
    k_init_u32<<<CDIV(N_ * 4, TPB), TPB, 0, stream>>>(amaxb, ENC_NEG_INF, N_ * 4);

    k_edge_alpha<<<CDIV(E4, TPB), TPB, 0, stream>>>(xl, xr, edge_attr, src, dst,
                                                    We_l, att_l, alpha, amaxb, E_);
    k_amax_fix<<<CDIV(N_ * 4, TPB), TPB, 0, stream>>>(amaxb, amax, N_ * 4);
    k_edge_den<<<CDIV(E4, TPB), TPB, 0, stream>>>(alpha, dst, amax, denom, E_);
    k_edge_scatter<<<CDIV(E4, TPB), TPB, 0, stream>>>(alpha, src, dst, denom, xl, agg, E_);

    k_col_stats<<<128, TPB, 0, stream>>>(agg, N_, 128, mean, istd);
    k_bn_elu_res<<<CDIV(NH, TPB), TPB, 0, stream>>>(agg, mean, istd,
                                                    bn_g + (size_t)l * 128,
                                                    bn_b + (size_t)l * 128,
                                                    xn, xnbf, NH);
  }

  // ---- 3-way global pooling ----
  hipMemsetAsync(sadd, 0, (size_t)G_ * 128 * 4, stream);
  hipMemsetAsync(cnt, 0, (size_t)G_ * 4, stream);
  k_init_u32<<<CDIV(G_ * 128, TPB), TPB, 0, stream>>>(smaxb, ENC_NEG_INF, G_ * 128);
  k_pool_scatter<<<CDIV(NH, TPB), TPB, 0, stream>>>(xn, batch, sadd, smaxb, cnt, N_);
  k_pool_fin<<<CDIV(G_ * 128, TPB), TPB, 0, stream>>>(sadd, smaxb, cnt, xgbf, G_);

  // ---- MLP tail (WMMA) ----
  k_pack_b<<<CDIV(384 * 256, TPB), TPB, 0, stream>>>(t1_W, packA, 384, 256);
  {
    dim3 grid(256 / 16, CDIV(G_, 16));
    k_gemm_wmma<<<grid, 32, 0, stream>>>(xgbf, packA, t1pre, G_, 384, 256);
  }
  k_col_stats<<<256, TPB, 0, stream>>>(t1pre, G_, 256, mean, istd);
  k_bn_relu<<<CDIV(G_ * 256, TPB), TPB, 0, stream>>>(t1pre, mean, istd, t1_g, t1_beta,
                                                     (float*)nullptr, t1bf, G_ * 256, 256);

  k_pack_b<<<CDIV(256 * 128, TPB), TPB, 0, stream>>>(t2_W, packB, 256, 128);
  {
    dim3 grid(128 / 16, CDIV(G_, 16));
    k_gemm_wmma<<<grid, 32, 0, stream>>>(t1bf, packB, t2pre, G_, 256, 128);
  }
  k_col_stats<<<128, TPB, 0, stream>>>(t2pre, G_, 128, mean, istd);
  k_bn_relu<<<CDIV(G_ * 128, TPB), TPB, 0, stream>>>(t2pre, mean, istd, t2_g, t2_beta,
                                                     t2act, t2bf, G_ * 128, 128);

  // ---- head + sigmoid ----
  k_head<<<CDIV(G_ * 12, TPB), TPB, 0, stream>>>(t2act, head_W, head_b, out, G_);
}
